// SetAbstraction_24524263260775
// MI455X (gfx1250) — matrix operand order repaired and checked by reading, not compiled, not run on hardware
//
#include <hip/hip_runtime.h>
#include <hip/hip_bf16.h>
#include <stdint.h>

#define B_   4
#define N_   8192
#define CF_  64      // input feature channels
#define S_   2048    // NPOINT
#define K_   32      // NSAMPLE
#define R2_  0.04f   // RADIUS^2
#define CIN_ 67      // 3 + 64
#define KP0_ 96      // layer0 K padded to 3x32
#define C0_  64
#define C1_  64
#define C2_  128

typedef __bf16 bf16x8 __attribute__((ext_vector_type(8)));
typedef __bf16 v16bf  __attribute__((ext_vector_type(16)));
typedef float  v8f    __attribute__((ext_vector_type(8)));

__device__ __forceinline__ unsigned short f2bf(float f) {
  unsigned u = __float_as_uint(f);
  unsigned r = (u + 0x7fffu + ((u >> 16) & 1u)) >> 16;  // round-to-nearest-even
  return (unsigned short)r;
}

__device__ __forceinline__ v8f wmma_bf16(v16bf a, v16bf b, v8f c) {
  return __builtin_amdgcn_wmma_f32_16x16x32_bf16(
      /*neg_a=*/false, a, /*neg_b=*/false, b,
      /*c_mod=*/(short)0, c, /*reuse_a=*/false, /*reuse_b=*/false);
}

// ---------------------------------------------------------------------------
// Kernel 1: fold BN into bf16 weights + f32 biases.
//   W'[o][c] = w[o][c] * gamma[o]*rsqrt(var[o]+eps)
//   b'[o]    = (b[o]-mu[o]) * gamma[o]*rsqrt(var[o]+eps) + beta[o]
// ---------------------------------------------------------------------------
__global__ void __launch_bounds__(256) prep_kernel(
    const float* __restrict__ w0, const float* __restrict__ b0,
    const float* __restrict__ g0, const float* __restrict__ be0,
    const float* __restrict__ m0, const float* __restrict__ v0,
    const float* __restrict__ w1, const float* __restrict__ b1,
    const float* __restrict__ g1, const float* __restrict__ be1,
    const float* __restrict__ m1, const float* __restrict__ v1,
    const float* __restrict__ w2, const float* __restrict__ b2,
    const float* __restrict__ g2, const float* __restrict__ be2,
    const float* __restrict__ m2, const float* __restrict__ v2,
    unsigned short* __restrict__ W0p, unsigned short* __restrict__ W1p,
    unsigned short* __restrict__ W2p,
    float* __restrict__ bias0, float* __restrict__ bias1, float* __restrict__ bias2) {
  const int tid = threadIdx.x;
  for (int o = tid; o < C0_; o += 256) {
    float s = g0[o] * rsqrtf(v0[o] + 1e-5f);
    bias0[o] = (b0[o] - m0[o]) * s + be0[o];
  }
  for (int o = tid; o < C1_; o += 256) {
    float s = g1[o] * rsqrtf(v1[o] + 1e-5f);
    bias1[o] = (b1[o] - m1[o]) * s + be1[o];
  }
  for (int o = tid; o < C2_; o += 256) {
    float s = g2[o] * rsqrtf(v2[o] + 1e-5f);
    bias2[o] = (b2[o] - m2[o]) * s + be2[o];
  }
  for (int i = tid; i < C0_ * KP0_; i += 256) {
    int o = i / KP0_, c = i % KP0_;
    float s = g0[o] * rsqrtf(v0[o] + 1e-5f);
    float w = (c < CIN_) ? w0[o * CIN_ + c] * s : 0.0f;
    W0p[i] = f2bf(w);
  }
  for (int i = tid; i < C1_ * C0_; i += 256) {
    int o = i / C0_, c = i % C0_;
    float s = g1[o] * rsqrtf(v1[o] + 1e-5f);
    W1p[i] = f2bf(w1[i] * s);
  }
  for (int i = tid; i < C2_ * C1_; i += 256) {
    int o = i / C1_, c = i % C1_;
    float s = g2[o] * rsqrtf(v2[o] + 1e-5f);
    W2p[i] = f2bf(w2[i] * s);
    (void)c;
  }
}

// ---------------------------------------------------------------------------
// Kernel 2: farthest point sampling. One 1024-thread block per batch.
// dist[] lives in LDS; two-level (shfl + LDS) argmax each iteration.
// Also writes new_xyz = xyz[fps_idx] directly to d_out.
// ---------------------------------------------------------------------------
__global__ void __launch_bounds__(1024) fps_kernel(
    const float* __restrict__ xyz, float* __restrict__ out_xyz) {
  const int b = blockIdx.x;
  const int tid = threadIdx.x;
  const int lane = tid & 31, wid = tid >> 5;
  __shared__ float dist[N_];
  __shared__ float wval[32];
  __shared__ int widx[32];
  __shared__ int cur;
  const float* xb = xyz + (size_t)b * N_ * 3;
  for (int i = tid; i < N_; i += 1024) dist[i] = 1e10f;
  if (tid == 0) {
    cur = 0;  // scan emits far_0 = 0 first
    out_xyz[((size_t)b * S_) * 3 + 0] = xb[0];
    out_xyz[((size_t)b * S_) * 3 + 1] = xb[1];
    out_xyz[((size_t)b * S_) * 3 + 2] = xb[2];
  }
  __syncthreads();
  for (int it = 1; it < S_; ++it) {
    const int ci = cur;
    const float cx = xb[ci * 3 + 0], cy = xb[ci * 3 + 1], cz = xb[ci * 3 + 2];
    float bv = -1.0f;
    int bi = 0;
    for (int i = tid; i < N_; i += 1024) {
      float dx = xb[i * 3 + 0] - cx;
      float dy = xb[i * 3 + 1] - cy;
      float dz = xb[i * 3 + 2] - cz;
      float d = dx * dx + dy * dy + dz * dz;
      float nd = fminf(dist[i], d);
      dist[i] = nd;
      if (nd > bv || (nd == bv && i < bi)) { bv = nd; bi = i; }
    }
    for (int off = 16; off; off >>= 1) {
      float ov = __shfl_down(bv, off, 32);
      int oi = __shfl_down(bi, off, 32);
      if (ov > bv || (ov == bv && oi < bi)) { bv = ov; bi = oi; }
    }
    if (lane == 0) { wval[wid] = bv; widx[wid] = bi; }
    __syncthreads();
    if (wid == 0) {
      bv = wval[lane];
      bi = widx[lane];
      for (int off = 16; off; off >>= 1) {
        float ov = __shfl_down(bv, off, 32);
        int oi = __shfl_down(bi, off, 32);
        if (ov > bv || (ov == bv && oi < bi)) { bv = ov; bi = oi; }
      }
      if (lane == 0) {
        cur = bi;
        out_xyz[((size_t)b * S_ + it) * 3 + 0] = xb[bi * 3 + 0];
        out_xyz[((size_t)b * S_ + it) * 3 + 1] = xb[bi * 3 + 1];
        out_xyz[((size_t)b * S_ + it) * 3 + 2] = xb[bi * 3 + 2];
      }
    }
    __syncthreads();
  }
}

// ---------------------------------------------------------------------------
// Kernel 3: ball query. One wave32 per query; ballot collects the first
// K_ in-radius indices in ascending index order (matches sorted gidx).
// ---------------------------------------------------------------------------
__global__ void __launch_bounds__(256) ballq_kernel(
    const float* __restrict__ xyz, const float* __restrict__ new_xyz,
    int* __restrict__ gidx) {
  const int lane = threadIdx.x & 31;
  const int gq = blockIdx.x * 8 + (threadIdx.x >> 5);
  const int b = gq >> 11;  // / S_
  const float* xb = xyz + (size_t)b * N_ * 3;
  const float qx = new_xyz[gq * 3 + 0];
  const float qy = new_xyz[gq * 3 + 1];
  const float qz = new_xyz[gq * 3 + 2];
  int* row = gidx + (size_t)gq * K_;
  int count = 0;
  int first = N_ - 1;
  bool have = false;
  for (int base = 0; base < N_ && count < K_; base += 32) {
    const int idx = base + lane;
    float dx = xb[idx * 3 + 0] - qx;
    float dy = xb[idx * 3 + 1] - qy;
    float dz = xb[idx * 3 + 2] - qz;
    bool in = (dx * dx + dy * dy + dz * dz) <= R2_;
    unsigned mask = (unsigned)__ballot(in);
    if (mask) {
      if (!have) { first = base + __builtin_ctz(mask); have = true; }
      int below = __popc(mask & ((1u << lane) - 1u));
      int pos = count + below;
      if (in && pos < K_) row[pos] = idx;
      count += __popc(mask);
    }
  }
  int cnt = count < K_ ? count : K_;
  int i = cnt + lane;
  if (i < K_) row[i] = first;
}

// ---------------------------------------------------------------------------
// Kernel 4: fused gather + 3-layer MLP (bf16 WMMA, f32 accum) + max over K.
// 256 threads / block, 8 waves; each wave owns a 16-row M-tile end-to-end.
// Block covers 128 rows = 4 full queries.
// ---------------------------------------------------------------------------
__global__ void __launch_bounds__(256) mlp_kernel(
    const float* __restrict__ xyz, const float* __restrict__ feat,
    const float* __restrict__ new_xyz, const int* __restrict__ gidx,
    const unsigned short* __restrict__ W0p, const unsigned short* __restrict__ W1p,
    const unsigned short* __restrict__ W2p,
    const float* __restrict__ bias0, const float* __restrict__ bias1,
    const float* __restrict__ bias2, float* __restrict__ out_feat) {
  __shared__ unsigned short A0[128 * KP0_];  // 24 KB  (layer0 inputs, bf16)
  __shared__ unsigned short A1[128 * 64];    // 16 KB  (layer0/1 outputs, bf16)
  __shared__ float maxbuf[8 * 128];          //  4 KB

  const int tid = threadIdx.x;
  const int lane = tid & 31, wave = tid >> 5;
  const int rowBase = blockIdx.x * 128;

  // ---- gather: rel-xyz (3) ++ features (64) ++ zero pad -> A0[128][96] bf16
  {
    const int r = tid >> 1, half = tid & 1;
    const int gr = rowBase + r;
    const int q = gr >> 5;   // global query
    const int b = q >> 11;   // / S_
    int gi = gidx[gr];
    if (gi > N_ - 1) gi = N_ - 1;
    const float* frow = feat + ((size_t)b * N_ + gi) * CF_;
    unsigned short* arow = A0 + r * KP0_;
    if (half == 0) {
      const float* prow = xyz + ((size_t)b * N_ + gi) * 3;
#pragma unroll
      for (int d = 0; d < 3; ++d) arow[d] = f2bf(prow[d] - new_xyz[q * 3 + d]);
      for (int c = 3; c < 48; ++c) arow[c] = f2bf(frow[c - 3]);
    } else {
      for (int c = 48; c < CIN_; ++c) arow[c] = f2bf(frow[c - 3]);
      for (int c = CIN_; c < KP0_; ++c) arow[c] = 0;
    }
  }
  __syncthreads();

  const int nlow = lane & 15;
  const int hiHalf = (lane >= 16) ? 1 : 0;
  const int mrow = wave * 16 + nlow;  // A-fragment row for this lane

  // A 16x32 bf16 fragment: lanes<16 hold K=0..15,
  // lanes>=16 hold K=16..31 of the 32-wide K step (matches loadB).
  auto loadA = [&](const unsigned short* base, int ld, int ks) -> v16bf {
    const int c0 = ks * 32 + (hiHalf ? 16 : 0);
    const int c1 = ks * 32 + (hiHalf ? 24 : 8);
    const unsigned short* p = base + mrow * ld;
    bf16x8 lo = *(const bf16x8*)(p + c0);
    bf16x8 hi = *(const bf16x8*)(p + c1);
    return __builtin_shufflevector(lo, hi, 0, 1, 2, 3, 4, 5, 6, 7,
                                   8, 9, 10, 11, 12, 13, 14, 15);
  };
  // B 32x16 bf16 fragment: lanes<16: N=lane, K=0..15; lanes>=16: N=lane-16,
  // K=16..31. Weights are [N][K] row-major (== reference w(o,c)).
  auto loadB = [&](const unsigned short* W, int ld, int nt, int ks) -> v16bf {
    const int n = nt * 16 + nlow;
    const int kb = ks * 32 + (hiHalf ? 16 : 0);
    return *(const v16bf*)(W + n * ld + kb);
  };

  v8f acc[4];

  // ---- Layer 0: 67(->96) -> 64
  {
    v16bf a0 = loadA(A0, KP0_, 0);
    v16bf a1 = loadA(A0, KP0_, 1);
    v16bf a2 = loadA(A0, KP0_, 2);
#pragma unroll
    for (int nt = 0; nt < 4; ++nt) {
      v8f c = {};
      c = wmma_bf16(a0, loadB(W0p, KP0_, nt, 0), c);
      c = wmma_bf16(a1, loadB(W0p, KP0_, nt, 1), c);
      c = wmma_bf16(a2, loadB(W0p, KP0_, nt, 2), c);
      acc[nt] = c;
    }
#pragma unroll
    for (int nt = 0; nt < 4; ++nt) {
      const int col = nt * 16 + nlow;
      const float bia = bias0[col];
#pragma unroll
      for (int j = 0; j < 8; ++j) {
        const int row = wave * 16 + j + (hiHalf ? 8 : 0);
        A1[row * 64 + col] = f2bf(fmaxf(acc[nt][j] + bia, 0.0f));
      }
    }
  }

  // ---- Layer 1: 64 -> 64 (read all of A1 for our rows, then overwrite)
  {
    v16bf a0 = loadA(A1, 64, 0);
    v16bf a1 = loadA(A1, 64, 1);
#pragma unroll
    for (int nt = 0; nt < 4; ++nt) {
      v8f c = {};
      c = wmma_bf16(a0, loadB(W1p, 64, nt, 0), c);
      c = wmma_bf16(a1, loadB(W1p, 64, nt, 1), c);
      acc[nt] = c;
    }
#pragma unroll
    for (int nt = 0; nt < 4; ++nt) {
      const int col = nt * 16 + nlow;
      const float bia = bias1[col];
#pragma unroll
      for (int j = 0; j < 8; ++j) {
        const int row = wave * 16 + j + (hiHalf ? 8 : 0);
        A1[row * 64 + col] = f2bf(fmaxf(acc[nt][j] + bia, 0.0f));
      }
    }
  }

  // ---- Layer 2: 64 -> 128, fused max over the 16 rows of this M-tile
  {
    v16bf a0 = loadA(A1, 64, 0);
    v16bf a1 = loadA(A1, 64, 1);
#pragma unroll
    for (int nt = 0; nt < 8; ++nt) {
      v8f c = {};
      c = wmma_bf16(a0, loadB(W2p, 64, nt, 0), c);
      c = wmma_bf16(a1, loadB(W2p, 64, nt, 1), c);
      const int col = nt * 16 + nlow;
      const float bia = bias2[col];
      float m = 0.0f;  // ReLU => outputs >= 0, so 0 is the identity
#pragma unroll
      for (int j = 0; j < 8; ++j) m = fmaxf(m, fmaxf(c[j] + bia, 0.0f));
      float o = __shfl_xor(m, 16, 32);  // combine rows j and j+8 halves
      m = fmaxf(m, o);
      if (!hiHalf) maxbuf[wave * 128 + col] = m;
    }
  }
  __syncthreads();

  // ---- combine wave pairs (two 16-row halves per query), write output
  for (int i = tid; i < 512; i += 256) {
    const int q = i >> 7, col = i & 127;
    float v = fmaxf(maxbuf[(q * 2) * 128 + col], maxbuf[(q * 2 + 1) * 128 + col]);
    const int gq = blockIdx.x * 4 + q;
    out_feat[(size_t)gq * C2_ + col] = v;
  }
}

// ---------------------------------------------------------------------------
extern "C" void kernel_launch(void* const* d_in, const int* in_sizes, int n_in,
                              void* d_out, int out_size, void* d_ws, size_t ws_size,
                              hipStream_t stream) {
  (void)in_sizes; (void)n_in; (void)out_size; (void)ws_size;
  const float* xyz  = (const float*)d_in[0];
  const float* feat = (const float*)d_in[1];
  const float* w0 = (const float*)d_in[2];
  const float* b0 = (const float*)d_in[3];
  const float* g0 = (const float*)d_in[4];
  const float* be0 = (const float*)d_in[5];
  const float* m0 = (const float*)d_in[6];
  const float* v0 = (const float*)d_in[7];
  const float* w1 = (const float*)d_in[8];
  const float* b1 = (const float*)d_in[9];
  const float* g1 = (const float*)d_in[10];
  const float* be1 = (const float*)d_in[11];
  const float* m1 = (const float*)d_in[12];
  const float* v1 = (const float*)d_in[13];
  const float* w2 = (const float*)d_in[14];
  const float* b2 = (const float*)d_in[15];
  const float* g2 = (const float*)d_in[16];
  const float* be2 = (const float*)d_in[17];
  const float* m2 = (const float*)d_in[18];
  const float* v2 = (const float*)d_in[19];

  char* ws = (char*)d_ws;
  int* gidx = (int*)ws;                                         // 1 MB
  unsigned short* W0p = (unsigned short*)(ws + (1 << 20));          // 12 KB
  unsigned short* W1p = (unsigned short*)(ws + (1 << 20) + 0x3000); //  8 KB
  unsigned short* W2p = (unsigned short*)(ws + (1 << 20) + 0x5000); // 16 KB
  float* bias0 = (float*)(ws + (1 << 20) + 0x9000);
  float* bias1 = bias0 + 64;
  float* bias2 = bias1 + 64;

  float* out_xyz = (float*)d_out;                 // (B,S,3)
  float* out_feat = out_xyz + (size_t)B_ * S_ * 3;  // (B,S,128)

  prep_kernel<<<1, 256, 0, stream>>>(w0, b0, g0, be0, m0, v0,
                                     w1, b1, g1, be1, m1, v1,
                                     w2, b2, g2, be2, m2, v2,
                                     W0p, W1p, W2p, bias0, bias1, bias2);
  fps_kernel<<<B_, 1024, 0, stream>>>(xyz, out_xyz);
  ballq_kernel<<<(B_ * S_) / 8, 256, 0, stream>>>(xyz, out_xyz, gidx);
  mlp_kernel<<<(B_ * S_ * K_) / 128, 256, 0, stream>>>(
      xyz, feat, out_xyz, gidx, W0p, W1p, W2p, bias0, bias1, bias2, out_feat);
}
